// FPCrossEncoderRankLearner_36919538876511
// MI455X (gfx1250) — matrix operand order repaired and checked by reading, not compiled, not run on hardware
//
#include <hip/hip_runtime.h>
#include <hip/hip_bf16.h>
#include <math.h>

// ---------------------------------------------------------------------------
// Problem constants (match reference)
// ---------------------------------------------------------------------------
#define B_   128
#define G_   64
#define T_   8192          // B_*G_
#define NB_  4096
#define DIN_ 12288         // 3*NB_
#define H1_  512
#define H2_  256
#define KSTEPS1 (DIN_/32)  // 384
#define KSTEPS2 (H1_/32)   // 16
#define LN_EPS 1e-5f

#define PANEL1_BYTES (32 * H1_ * 2)   // 32768: one k-step B panel for GEMM1
#define PANEL1_F16   (32 * H1_)       // 16384 f16 elements
#define QUARTER_BYTES (PANEL1_BYTES / 4)  // 8192: per-wave TDM slice
#define QUARTER_F16   (PANEL1_F16 / 4)    // 4096

typedef __attribute__((ext_vector_type(16))) _Float16 v16h;
typedef __attribute__((ext_vector_type(8)))  _Float16 f16x8;
typedef __attribute__((ext_vector_type(8)))  float    v8f;
typedef __attribute__((ext_vector_type(4)))  unsigned int v4u;
typedef __attribute__((ext_vector_type(8)))  int      v8i;
typedef __attribute__((ext_vector_type(4)))  int      v4i;

#if defined(__has_builtin)
#if __has_builtin(__builtin_amdgcn_tensor_load_to_lds)
#define HAVE_TDM 1
#endif
#endif

// ---------------------------------------------------------------------------
// Prep kernels: f16 materialization + WMMA-fragment-order weight packing
// ---------------------------------------------------------------------------
__global__ void k_sigmoid_f16(const float* __restrict__ logits,
                              _Float16* __restrict__ P16, int n) {
    int i = blockIdx.x * blockDim.x + threadIdx.x;
    if (i < n) P16[i] = (_Float16)(1.0f / (1.0f + expf(-logits[i])));
}

// C16 = (f16)cand, PC16 = preds[b] * cand  (t -> b = t>>6, NB=4096 -> shift 12)
__global__ void k_cand_f16(const int* __restrict__ cand,
                           const _Float16* __restrict__ P16,
                           _Float16* __restrict__ C16,
                           _Float16* __restrict__ PC16) {
    size_t i = (size_t)blockIdx.x * blockDim.x + threadIdx.x;   // < T_*NB_
    int t   = (int)(i >> 12);
    int col = (int)(i & (NB_ - 1));
    int b   = t >> 6;
    float c = (float)cand[i];
    float p = (float)P16[((size_t)b << 12) + col];
    C16[i]  = (_Float16)c;
    PC16[i] = (_Float16)(p * c);
}

// Pack W1 [DIN_,H1_] f32 -> W1p, panel-contiguous fragment order:
//   W1p[((kstep*32 + ntile)*32 + lane)*16 + j]
//   per lane: col = ntile*16 + (lane&15); K = kstep*32 + ((lane>=16)?8:0) + (j&7) + ((j>=8)?16:0)
// => each k-step's 32 KB B panel is one contiguous block (TDM-friendly).
__global__ void k_packW1(const float* __restrict__ W1, _Float16* __restrict__ W1p) {
    size_t idx = (size_t)blockIdx.x * blockDim.x + threadIdx.x; // < 384*32*32*16
    int j    = (int)(idx & 15);
    int lane = (int)((idx >> 4) & 31);
    int rest = (int)(idx >> 9);
    int ntile = rest & 31;
    int kstep = rest >> 5;
    int k = kstep * 32 + ((lane >> 4) << 3) + (j & 7) + ((j >> 3) << 4);
    int n = ntile * 16 + (lane & 15);
    W1p[idx] = (_Float16)W1[(size_t)k * H1_ + n];
}

// W2p stays in [ntile][kstep][lane][16] order (direct-load path)
__global__ void k_packW2(const float* __restrict__ W2, _Float16* __restrict__ W2p) {
    size_t idx = (size_t)blockIdx.x * blockDim.x + threadIdx.x; // < 16*16*32*16
    int j    = (int)(idx & 15);
    int lane = (int)((idx >> 4) & 31);
    int rest = (int)(idx >> 9);
    int kstep = rest & (KSTEPS2 - 1);
    int ntile = rest >> 4;
    int k = kstep * 32 + ((lane >> 4) << 3) + (j & 7) + ((j >> 3) << 4);
    int n = ntile * 16 + (lane & 15);
    W2p[idx] = (_Float16)W2[(size_t)k * H2_ + n];
}

__device__ __forceinline__ float gelu_exact(float x) {
    return 0.5f * x * (1.0f + erff(x * 0.70710678118654752f));
}

__device__ __forceinline__ v16h load_afrag(const _Float16* p) {
    f16x8 lo = *(const f16x8*)(p);
    f16x8 hi = *(const f16x8*)(p + 16);
    return __builtin_shufflevector(lo, hi, 0,1,2,3,4,5,6,7,8,9,10,11,12,13,14,15);
}

#ifdef HAVE_TDM
// 1-D TDM copy: nbytes (dword multiple) from global -> LDS offset.
// D# per CDNA5 ISA ch.8: group0 = {count|flags, lds_addr, gaddr_lo, gaddr_hi|type=2}
// group1: data_size=4B; tensor_dim0 = tile_dim0 = ndwords; tensor_dim1 = 1.
__device__ __forceinline__ void tdm_load_1d(unsigned lds_off, const void* gsrc,
                                            unsigned nbytes) {
    unsigned long long ga = (unsigned long long)(size_t)gsrc;
    unsigned ndw = nbytes >> 2;
    v4u g0 = {0u, 0u, 0u, 0u};
    g0[0] = 1u;                                        // count=1 (valid D#)
    g0[1] = lds_off;                                   // lds_addr (bytes)
    g0[2] = (unsigned)(ga & 0xFFFFFFFFu);              // global_addr[31:0]
    g0[3] = (unsigned)((ga >> 32) & 0x1FFFFFFu)        // global_addr[56:32]
          | (2u << 30);                                // type=2 ("image")
    v8i g1 = {0, 0, 0, 0, 0, 0, 0, 0};
    g1[0] = (2 << 16);                                 // data_size = 4 bytes
    g1[1] = (int)((ndw & 0xFFFFu) << 16);              // tensor_dim0[15:0]
    g1[2] = (int)(((ndw >> 16) & 0xFFFFu) | (1u << 16)); // dim0 hi | tensor_dim1=1
    g1[3] = (int)((ndw & 0xFFFFu) << 16);              // tile_dim0 (<=65535 dwords)
    v4i z4 = {0, 0, 0, 0};
#if __has_include(<hip/amd_detail/amd_gfx1250_TDM.h>)
    v8i z8 = {0, 0, 0, 0, 0, 0, 0, 0};
    __builtin_amdgcn_tensor_load_to_lds(g0, g1, z4, z4, z8, 0);
#else
    __builtin_amdgcn_tensor_load_to_lds(g0, g1, z4, z4, 0);
#endif
}
#endif

// ---------------------------------------------------------------------------
// GEMM1: H1pre = X @ W1 + b1 -> GELU -> LayerNorm -> H1buf (f16)
// Block = 256 threads = 8 waves; 32 rows x all 512 cols per block.
// B panels stream global->LDS via TDM: waves 0..3 each DMA one 8 KB quarter
// of the next panel (4 concurrent descriptors), TENSORcnt waits are per-wave
// and unconditional (no-op for non-issuing waves); workgroup barrier releases.
// All 16 ds_load_b128 for a k-step issue before the 8-WMMA chain.
// ---------------------------------------------------------------------------
__launch_bounds__(256)
__global__ void k_gemm1(const _Float16* __restrict__ P16,
                        const _Float16* __restrict__ C16,
                        const _Float16* __restrict__ PC16,
                        const _Float16* __restrict__ W1p,
                        const float* __restrict__ b1,
                        const float* __restrict__ g1,
                        const float* __restrict__ be1,
                        _Float16* __restrict__ H1buf) {
    // 2 x 32KB B-panel staging; epilogue aliases [0,32KB) + stats at +32KB.
    __shared__ __align__(32) char ldsbuf[2 * PANEL1_BYTES];   // 64 KB total

    const int tid    = threadIdx.x;
    const int lane   = tid & 31;
    const int w      = tid >> 5;
    const int m_half = w >> 2;
    const int n_block = w & 3;
    const int m0     = blockIdx.x * 32;
    const int grp    = m0 >> 6;                        // group id (preds row)
    const int row_g  = m0 + m_half * 16 + (lane & 15); // per-lane A row
    const int klo    = (lane >> 4) << 3;               // 0 or 8

    v8f acc[8];
    #pragma unroll
    for (int i = 0; i < 8; ++i) acc[i] = (v8f){0,0,0,0,0,0,0,0};

    const _Float16* predrow = P16  + (size_t)grp  * NB_;
    const _Float16* candrow = C16  + (size_t)row_g * NB_;
    const _Float16* pcrow   = PC16 + (size_t)row_g * NB_;

#ifdef HAVE_TDM
    const unsigned lds_base = (unsigned)(size_t)(void*)&ldsbuf[0];
    // Preload panel 0: waves 0..3 DMA one quarter each
    if (w < 4) {
        tdm_load_1d(lds_base + (unsigned)w * QUARTER_BYTES,
                    W1p + (size_t)w * QUARTER_F16, QUARTER_BYTES);
    }
    __builtin_amdgcn_s_wait_tensorcnt(0);
    __syncthreads();
#else
    {   // cooperative copy fallback: 256 threads x 4 x 32B = 32 KB
        const v16h* src = (const v16h*)W1p;
        v16h* dst = (v16h*)&ldsbuf[0];
        #pragma unroll
        for (int c = 0; c < 4; ++c) dst[tid + c * 256] = src[tid + c * 256];
        __syncthreads();
    }
#endif

    for (int kstep = 0; kstep < KSTEPS1; ++kstep) {
        const int cur = kstep & 1;

        // Kick off async TDM quarters of the next panel into the other buffer
#ifdef HAVE_TDM
        if (w < 4 && (kstep + 1) < KSTEPS1) {
            tdm_load_1d(lds_base + (unsigned)((cur ^ 1) * PANEL1_BYTES)
                                 + (unsigned)w * QUARTER_BYTES,
                        W1p + (size_t)(kstep + 1) * PANEL1_F16
                            + (size_t)w * QUARTER_F16,
                        QUARTER_BYTES);
        }
#endif

        // A fragment (on-the-fly X construction; preds row shared by group)
        const int seg = kstep >> 7;                    // 128 ksteps / segment
        const int kk  = (kstep & 127) << 5;
        const _Float16* arow = (seg == 0) ? predrow : (seg == 1) ? candrow : pcrow;
        __builtin_prefetch(arow + kk + 256, 0, 3);
        v16h afrag = load_afrag(arow + kk + klo);

        // Load all 8 B fragments first (16 ds_load_b128 in flight), then WMMA.
        const _Float16* lpanel = (const _Float16*)&ldsbuf[cur * PANEL1_BYTES];
        v16h bfrag[8];
        #pragma unroll
        for (int i = 0; i < 8; ++i) {
            bfrag[i] = *(const v16h*)
                (lpanel + ((size_t)(n_block * 8 + i) * 32 + lane) * 16);
        }
        #pragma unroll
        for (int i = 0; i < 8; ++i) {
            acc[i] = __builtin_amdgcn_wmma_f32_16x16x32_f16(
                false, afrag, false, bfrag[i], (short)0, acc[i], false, false);
        }

#ifdef HAVE_TDM
        __builtin_amdgcn_s_wait_tensorcnt(0);          // own quarter landed
        __syncthreads();                               // all quarters visible
#else
        __syncthreads();
        if ((kstep + 1) < KSTEPS1) {
            const v16h* src = (const v16h*)(W1p + (size_t)(kstep + 1) * PANEL1_F16);
            v16h* dst = (v16h*)&ldsbuf[(cur ^ 1) * PANEL1_BYTES];
            #pragma unroll
            for (int c = 0; c < 4; ++c) dst[tid + c * 256] = src[tid + c * 256];
            __syncthreads();
        }
#endif
    }

    // Epilogue: bias + GELU -> LDS f16 (aliases staging buffer 0)
    _Float16* sh = (_Float16*)&ldsbuf[0];              // 32 rows x 512 f16
    float* smu = (float*)&ldsbuf[PANEL1_BYTES];        // 32 f32
    float* srs = (float*)&ldsbuf[PANEL1_BYTES + 128];  // 32 f32

    const int rowoff = m_half * 16 + ((lane >> 4) << 3);
    const int coloff = n_block * 128 + (lane & 15);
    #pragma unroll
    for (int i = 0; i < 8; ++i) {
        int col = coloff + i * 16;
        float bb = b1[col];
        #pragma unroll
        for (int r = 0; r < 8; ++r) {
            float x = acc[i][r] + bb;
            sh[(rowoff + r) * H1_ + col] = (_Float16)gelu_exact(x);
        }
    }
    __syncthreads();

    if (tid < 32) {
        float s = 0.f, s2 = 0.f;
        for (int c = 0; c < H1_; ++c) {
            float v = (float)sh[tid * H1_ + c];
            s += v; s2 += v * v;
        }
        float mu  = s * (1.0f / H1_);
        float var = s2 * (1.0f / H1_) - mu * mu;
        smu[tid] = mu;
        srs[tid] = rsqrtf(var + LN_EPS);
    }
    __syncthreads();

    for (int idx = tid; idx < 32 * H1_; idx += 256) {
        int r = idx >> 9, c = idx & (H1_ - 1);
        float v = (float)sh[idx];
        float o = (v - smu[r]) * srs[r] * g1[c] + be1[c];
        H1buf[(size_t)(m0 + r) * H1_ + c] = (_Float16)o;
    }
}

// ---------------------------------------------------------------------------
// GEMM2: H2 = LN(GELU(H1buf @ W2 + b2)) -> H2buf (f16). K-loop 16, 4 tiles/wave.
// ---------------------------------------------------------------------------
__launch_bounds__(256)
__global__ void k_gemm2(const _Float16* __restrict__ H1buf,
                        const _Float16* __restrict__ W2p,
                        const float* __restrict__ b2,
                        const float* __restrict__ g2,
                        const float* __restrict__ be2,
                        _Float16* __restrict__ H2buf) {
    __shared__ _Float16 sh[32 * H2_];        // 16 KB
    __shared__ float smu[32], srs[32];

    const int tid    = threadIdx.x;
    const int lane   = tid & 31;
    const int w      = tid >> 5;
    const int m_half = w >> 2;
    const int n_block = w & 3;
    const int m0     = blockIdx.x * 32;
    const int row_g  = m0 + m_half * 16 + (lane & 15);
    const int klo    = (lane >> 4) << 3;

    v8f acc[4];
    #pragma unroll
    for (int i = 0; i < 4; ++i) acc[i] = (v8f){0,0,0,0,0,0,0,0};

    const _Float16* arow = H1buf + (size_t)row_g * H1_;

    for (int kstep = 0; kstep < KSTEPS2; ++kstep) {
        const int kk = kstep << 5;
        v16h afrag = load_afrag(arow + kk + klo);
        const _Float16* bbase =
            W2p + ((size_t)((n_block * 4) * KSTEPS2 + kstep) * 32 + lane) * 16;
        v16h bfrag[4];
        #pragma unroll
        for (int i = 0; i < 4; ++i)
            bfrag[i] = *(const v16h*)(bbase + (size_t)i * KSTEPS2 * 32 * 16);
        #pragma unroll
        for (int i = 0; i < 4; ++i) {
            acc[i] = __builtin_amdgcn_wmma_f32_16x16x32_f16(
                false, afrag, false, bfrag[i], (short)0, acc[i], false, false);
        }
    }

    const int rowoff = m_half * 16 + ((lane >> 4) << 3);
    const int coloff = n_block * 64 + (lane & 15);
    #pragma unroll
    for (int i = 0; i < 4; ++i) {
        int col = coloff + i * 16;
        float bb = b2[col];
        #pragma unroll
        for (int r = 0; r < 8; ++r) {
            float x = acc[i][r] + bb;
            sh[(rowoff + r) * H2_ + col] = (_Float16)gelu_exact(x);
        }
    }
    __syncthreads();

    if (tid < 32) {
        float s = 0.f, s2 = 0.f;
        for (int c = 0; c < H2_; ++c) {
            float v = (float)sh[tid * H2_ + c];
            s += v; s2 += v * v;
        }
        float mu  = s * (1.0f / H2_);
        float var = s2 * (1.0f / H2_) - mu * mu;
        smu[tid] = mu;
        srs[tid] = rsqrtf(var + LN_EPS);
    }
    __syncthreads();

    for (int idx = tid; idx < 32 * H2_; idx += 256) {
        int r = idx >> 8, c = idx & (H2_ - 1);
        float v = (float)sh[idx];
        float o = (v - smu[r]) * srs[r] * g2[c] + be2[c];
        H2buf[(size_t)(m0 + r) * H2_ + c] = (_Float16)o;
    }
}

// ---------------------------------------------------------------------------
// scores[t] = H2buf[t,:] . W3 + b3
// ---------------------------------------------------------------------------
__global__ void k_scores(const _Float16* __restrict__ H2buf,
                         const float* __restrict__ W3,
                         const float* __restrict__ b3,
                         float* __restrict__ scores) {
    int t = blockIdx.x * blockDim.x + threadIdx.x;   // < T_
    const _Float16* h = H2buf + (size_t)t * H2_;
    float sum = 0.f;
    #pragma unroll 8
    for (int j = 0; j < H2_; ++j) sum += (float)h[j] * W3[j];
    scores[t] = sum + b3[0];
}

// ---------------------------------------------------------------------------
// Loss: per group (positive is index 0; mask covers all 64 entries):
//   loss_b = logsumexp(s/T) - s0/T;  out = mean over B groups.
// ---------------------------------------------------------------------------
__global__ void k_loss(const float* __restrict__ scores, float* __restrict__ out) {
    __shared__ float red[B_];
    const int b = threadIdx.x;                       // 128 threads
    const float it = 10.0f;                          // 1/TEMP
    const float* s = scores + b * G_;
    float m = -1e30f;
    for (int j = 0; j < G_; ++j) m = fmaxf(m, s[j] * it);
    float sum = 0.f;
    for (int j = 0; j < G_; ++j) sum += expf(s[j] * it - m);
    float lse = m + logf(sum);
    red[b] = lse - s[0] * it;
    __syncthreads();
    for (int off = B_ / 2; off > 0; off >>= 1) {
        if (b < off) red[b] += red[b + off];
        __syncthreads();
    }
    if (b == 0) out[0] = red[0] * (1.0f / B_);
}

// ---------------------------------------------------------------------------
// Host launcher
// ---------------------------------------------------------------------------
extern "C" void kernel_launch(void* const* d_in, const int* in_sizes, int n_in,
                              void* d_out, int out_size, void* d_ws, size_t ws_size,
                              hipStream_t stream) {
    const float* logits  = (const float*)d_in[0];
    // d_in[1] true_fp unused; d_in[3] batch_ptr uniform G=64; d_in[4] labels structural
    const int*   cand_fp = (const int*)d_in[2];
    const float* W1 = (const float*)d_in[5];
    const float* b1 = (const float*)d_in[6];
    const float* g1 = (const float*)d_in[7];
    const float* be1= (const float*)d_in[8];
    const float* W2 = (const float*)d_in[9];
    const float* b2 = (const float*)d_in[10];
    const float* g2 = (const float*)d_in[11];
    const float* be2= (const float*)d_in[12];
    const float* W3 = (const float*)d_in[13];
    const float* b3 = (const float*)d_in[14];
    float* out = (float*)d_out;

    // Workspace layout (bytes)
    char* ws = (char*)d_ws;
    size_t off = 0;
    _Float16* P16  = (_Float16*)(ws + off); off += (size_t)B_ * NB_ * 2;   // 1 MB
    _Float16* C16  = (_Float16*)(ws + off); off += (size_t)T_ * NB_ * 2;   // 64 MB
    _Float16* PC16 = (_Float16*)(ws + off); off += (size_t)T_ * NB_ * 2;   // 64 MB
    _Float16* W1p  = (_Float16*)(ws + off); off += (size_t)DIN_ * H1_ * 2; // 12.6 MB
    _Float16* W2p  = (_Float16*)(ws + off); off += (size_t)H1_ * H2_ * 2;  // 0.25 MB
    _Float16* H1b  = (_Float16*)(ws + off); off += (size_t)T_ * H1_ * 2;   // 8 MB
    _Float16* H2b  = (_Float16*)(ws + off); off += (size_t)T_ * H2_ * 2;   // 4 MB
    float*    sc   = (float*)(ws + off);    off += (size_t)T_ * 4;         // 32 KB

    // 1) preds = sigmoid(logits) -> f16
    {
        int n = B_ * NB_;
        k_sigmoid_f16<<<(n + 255) / 256, 256, 0, stream>>>(logits, P16, n);
    }
    // 2) cand -> f16, preds*cand -> f16  (single HBM pass over cand_fp)
    {
        size_t n = (size_t)T_ * NB_;
        k_cand_f16<<<(unsigned)(n / 256), 256, 0, stream>>>(cand_fp, P16, C16, PC16);
    }
    // 3) pack weights into WMMA fragment order (W1: TDM panel-contiguous)
    {
        size_t n1 = (size_t)DIN_ * H1_;
        k_packW1<<<(unsigned)(n1 / 256), 256, 0, stream>>>(W1, W1p);
        size_t n2 = (size_t)H1_ * H2_;
        k_packW2<<<(unsigned)(n2 / 256), 256, 0, stream>>>(W2, W2p);
    }
    // 4) GEMM1 + GELU + LN  (TDM double-buffered B panels, L2-resident A)
    k_gemm1<<<T_ / 32, 256, 0, stream>>>(P16, C16, PC16, W1p, b1, g1, be1, H1b);
    // 5) GEMM2 + GELU + LN
    k_gemm2<<<T_ / 32, 256, 0, stream>>>(H1b, W2p, b2, g2, be2, H2b);
    // 6) scores
    k_scores<<<T_ / 256, 256, 0, stream>>>(H2b, W3, b3, sc);
    // 7) listwise loss -> scalar
    k_loss<<<1, B_, 0, stream>>>(sc, out);

    (void)in_sizes; (void)n_in; (void)out_size; (void)ws_size;
}
